// SingleMambaBlock_15977278341330
// MI455X (gfx1250) — compile-verified, matrix operand order checked by reading, not run on hardware
//
#include <hip/hip_runtime.h>

#define D_MODEL 256
#define D_STATE 16
#define DT_RANK 16
#define D_CONV  4
#define SEQ     4096
#define BATCH   2
#define BS      (BATCH*SEQ)     /* 8192 rows */
#define NCHUNK  64
#define CHLEN   (SEQ/NCHUNK)    /* 64 steps per chunk */
#define TPB     256

typedef __attribute__((ext_vector_type(16))) _Float16 v16h;
typedef __attribute__((ext_vector_type(8)))  _Float16 v8h;
typedef __attribute__((ext_vector_type(8)))  float    v8f;
typedef __attribute__((ext_vector_type(4)))  unsigned int v4u;
typedef __attribute__((ext_vector_type(4)))  int v4i;
typedef __attribute__((ext_vector_type(8)))  int v8i;

// ---------------------------------------------------------------- utilities
__device__ __forceinline__ float warp_sum32(float v) {
  v += __shfl_xor(v, 1, 32);
  v += __shfl_xor(v, 2, 32);
  v += __shfl_xor(v, 4, 32);
  v += __shfl_xor(v, 8, 32);
  v += __shfl_xor(v, 16, 32);
  return v;
}
__device__ __forceinline__ float half_sum16(float v) {
  // sums within each 16-lane half of a wave32
  v += __shfl_xor(v, 1, 32);
  v += __shfl_xor(v, 2, 32);
  v += __shfl_xor(v, 4, 32);
  v += __shfl_xor(v, 8, 32);
  return v;
}
__device__ __forceinline__ float silu_f(float x) {
  return x / (1.f + __expf(-x));
}
__device__ __forceinline__ float softplus_f(float x) {
  return (x > 20.f) ? x : log1pf(__expf(x));
}
__device__ __forceinline__ size_t chunk_idx(int dir, int b, int d, int c, int n) {
  return ((size_t)(((dir * BATCH + b) * D_MODEL + d) * NCHUNK + c)) * 16 + n;
}
__device__ __forceinline__ v16h frag16(const _Float16* p) {
  // ISA 16-bit fragment: per lane, 8 halfs at K=kh..kh+7 then 8 at K=kh+16..kh+23
  return __builtin_shufflevector(*(const v8h*)p, *(const v8h*)(p + 16),
                                 0,1,2,3,4,5,6,7,8,9,10,11,12,13,14,15);
}

// ---------------------------------------------------------------- TDM: 2D f16 panel -> LDS
// D# per CDNA5 ISA ch.8: group0 = {count/type/lds_addr/global_addr},
// group1 = {data_size, tensor_dim0/1, tile_dim0/1, tensor_dim0_stride}.
__device__ __forceinline__ void tdm_load_f16_2d(unsigned lds_off, const void* gptr,
                                                unsigned rows_total, unsigned rows_tile,
                                                unsigned k_elems) {
  unsigned long long ga = (unsigned long long)gptr;
  v4u g0;
  g0[0] = 1u;                                        // count=1 (valid user descriptor)
  g0[1] = lds_off;                                   // lds_addr (bytes)
  g0[2] = (unsigned)(ga & 0xffffffffu);              // global_addr[31:0]
  g0[3] = (unsigned)((ga >> 32) & 0x01ffffffu) | (2u << 30);  // addr[56:32] | type=2
  v8i g1;
  g1[0] = (int)(1u << 16);                           // data_size=1 (2 bytes); wg_mask=0
  g1[1] = (int)((k_elems & 0xffffu) << 16);          // tensor_dim0[15:0]
  g1[2] = (int)((k_elems >> 16) | ((rows_total & 0xffffu) << 16)); // dim0[31:16]|dim1[15:0]
  g1[3] = (int)((rows_total >> 16) | ((k_elems & 0xffffu) << 16)); // dim1[31:16]|tile_dim0
  g1[4] = (int)(rows_tile & 0xffffu);                // tile_dim1; tile_dim2=0
  g1[5] = (int)k_elems;                              // tensor_dim0_stride[31:0]
  g1[6] = 0;                                         // stride0[47:32] | stride1[15:0]
  g1[7] = 0;
  v4i gz = {0, 0, 0, 0};
#if defined(__clang_major__) && (__clang_major__ >= 23)
  v8i gz8 = {0, 0, 0, 0, 0, 0, 0, 0};
  __builtin_amdgcn_tensor_load_to_lds(g0, g1, gz, gz, gz8, 0);
#else
  __builtin_amdgcn_tensor_load_to_lds(g0, g1, gz, gz, 0);
#endif
}

// ---------------------------------------------------------------- f32 -> f16
__global__ void k_cvt_f2h(const float* __restrict__ a, _Float16* __restrict__ o, int n) {
  int i = blockIdx.x * blockDim.x + threadIdx.x;
  if (i < n) o[i] = (_Float16)a[i];
}

// ------------------------------------------------- LayerNorm (wave per row) -> f16
__global__ void k_ln_f16(const float* __restrict__ x, const float* __restrict__ g,
                         const float* __restrict__ be, _Float16* __restrict__ out, int rows) {
  int wid  = (blockIdx.x * blockDim.x + threadIdx.x) >> 5;
  int lane = threadIdx.x & 31;
  if (wid >= rows) return;
  const float* xr = x + (size_t)wid * D_MODEL;
  float v[8]; float s = 0.f, s2 = 0.f;
#pragma unroll
  for (int i = 0; i < 8; ++i) { float t = xr[lane * 8 + i]; v[i] = t; s += t; s2 += t * t; }
  s = warp_sum32(s); s2 = warp_sum32(s2);
  float mean = s * (1.f / D_MODEL);
  float var  = s2 * (1.f / D_MODEL) - mean * mean;
  float rstd = rsqrtf(var + 1e-5f);
  _Float16* orow = out + (size_t)wid * D_MODEL;
#pragma unroll
  for (int i = 0; i < 8; ++i) {
    int c = lane * 8 + i;
    orow[c] = (_Float16)((v[i] - mean) * rstd * g[c] + be[c]);
  }
}

// ------------------------------------------------- big WMMA GEMM, K==256, N multiple of 64
// Workgroup: 128(M) x 64(N) macro-tile; 8 waves each own 32x32 (2x2 register blocking).
// B panel (64 x 256 f16 = 32 KB) staged into LDS once per workgroup by the TDM.
__global__ void __launch_bounds__(TPB)
k_gemm_wmma_big(const _Float16* __restrict__ A, const _Float16* __restrict__ Bw,
                float* __restrict__ C, int M, int N) {
  __shared__ _Float16 lds_b[64 * 256];
  int nblk = N >> 6;
  int bm = blockIdx.x / nblk;
  int bn = blockIdx.x - bm * nblk;
  int m0 = bm * 128;
  int n0 = bn * 64;

  if ((threadIdx.x >> 5) == 0) {          // wave 0 issues the tensor DMA
    tdm_load_f16_2d((unsigned)(size_t)&lds_b[0], Bw + (size_t)n0 * 256,
                    (unsigned)(N - n0), 64u, 256u);
    __builtin_amdgcn_s_wait_tensorcnt(0);
  }
  __syncthreads();

  int wave = threadIdx.x >> 5;
  int lane = threadIdx.x & 31;
  int wm = wave >> 1;            // 0..3 -> M sub-block
  int wn = wave & 1;             // 0..1 -> N sub-block
  int l15 = lane & 15;
  int kh = (lane >> 4) << 3;     // 0 or 8

  const _Float16* ap0 = A + (size_t)(m0 + wm * 32 + l15) * 256 + kh;
  const _Float16* ap1 = ap0 + 16 * 256;
  const _Float16* bp0 = &lds_b[(size_t)(wn * 32 + l15) * 256 + kh];
  const _Float16* bp1 = bp0 + 16 * 256;

  v8f acc00 = {}, acc01 = {}, acc10 = {}, acc11 = {};
#pragma unroll
  for (int k0 = 0; k0 < 256; k0 += 32) {
    v16h a0 = frag16(ap0 + k0);
    v16h a1 = frag16(ap1 + k0);
    v16h b0 = frag16(bp0 + k0);
    v16h b1 = frag16(bp1 + k0);
    acc00 = __builtin_amdgcn_wmma_f32_16x16x32_f16(false, a0, false, b0, (short)0, acc00, false, false);
    acc01 = __builtin_amdgcn_wmma_f32_16x16x32_f16(false, a0, false, b1, (short)0, acc01, false, false);
    acc10 = __builtin_amdgcn_wmma_f32_16x16x32_f16(false, a1, false, b0, (short)0, acc10, false, false);
    acc11 = __builtin_amdgcn_wmma_f32_16x16x32_f16(false, a1, false, b1, (short)0, acc11, false, false);
  }
  // C/D layout: VGPR j -> M=j (lanes 0..15) / M=8+j (lanes 16..31); N = lane&15
  int crow = m0 + wm * 32 + ((lane >> 4) << 3);
  int ccol = n0 + wn * 32 + l15;
#pragma unroll
  for (int j = 0; j < 8; ++j) {
    C[(size_t)(crow + j)      * N + ccol]      = acc00[j];
    C[(size_t)(crow + j)      * N + ccol + 16] = acc01[j];
    C[(size_t)(crow + 16 + j) * N + ccol]      = acc10[j];
    C[(size_t)(crow + 16 + j) * N + ccol + 16] = acc11[j];
  }
}

// ------------------------------------------------- small WMMA GEMM (N=48): wave per 16x16 tile
__global__ void k_gemm_wmma_small(const _Float16* __restrict__ A, const _Float16* __restrict__ Bw,
                                  float* __restrict__ C, int M, int N, int K) {
  int wid  = (blockIdx.x * blockDim.x + threadIdx.x) >> 5;
  int lane = threadIdx.x & 31;
  int ntile = N >> 4;
  int tm = wid / ntile;
  int tn = wid - tm * ntile;
  if (tm * 16 >= M) return;
  int arow = tm * 16 + (lane & 15);
  int bcol = tn * 16 + (lane & 15);
  int kh   = (lane >> 4) << 3;
  const _Float16* ap = A  + (size_t)arow * K + kh;
  const _Float16* bp = Bw + (size_t)bcol * K + kh;
  v8f acc = {};
  for (int k0 = 0; k0 < K; k0 += 32) {
    v16h a = frag16(ap + k0);
    v16h b = frag16(bp + k0);
    acc = __builtin_amdgcn_wmma_f32_16x16x32_f16(false, a, false, b, (short)0, acc, false, false);
  }
  int crow0 = tm * 16 + ((lane >> 4) << 3);
  int ccol  = tn * 16 + (lane & 15);
#pragma unroll
  for (int j = 0; j < 8; ++j)
    C[(size_t)(crow0 + j) * N + ccol] = acc[j];
}

// ------------------------------------------------- depthwise conv (causal fwd, anti-causal bwd) + SiLU
__global__ void k_conv_silu(const float* __restrict__ xz,
                            const float* __restrict__ cwf, const float* __restrict__ cbf,
                            const float* __restrict__ cwb, const float* __restrict__ cbb,
                            float* __restrict__ uf, float* __restrict__ ub,
                            _Float16* __restrict__ ufh, _Float16* __restrict__ ubh) {
  int idx = blockIdx.x * blockDim.x + threadIdx.x;
  if (idx >= BS * D_MODEL) return;
  int d   = idx & (D_MODEL - 1);
  int row = idx >> 8;
  int t   = row & (SEQ - 1);
  float accf = cbf[d], accb = cbb[d];
#pragma unroll
  for (int k = 0; k < D_CONV; ++k) {
    int tf = t - (D_CONV - 1) + k;
    if (tf >= 0)
      accf += cwf[d * D_CONV + k] * xz[(size_t)(row - (D_CONV - 1) + k) * 512 + d];
    int tb = t + (D_CONV - 1) - k;
    if (tb < SEQ)
      accb += cwb[d * D_CONV + k] * xz[(size_t)(row + (D_CONV - 1) - k) * 512 + d];
  }
  accf = silu_f(accf); accb = silu_f(accb);
  uf[idx] = accf;  ub[idx] = accb;
  ufh[idx] = (_Float16)accf;  ubh[idx] = (_Float16)accb;
}

// ------------------------------------------------- dt = softplus(dbl[:, :16] @ dtW^T + bias)
__global__ void k_dtproj(const float* __restrict__ dbl, const float* __restrict__ w,
                         const float* __restrict__ bias, float* __restrict__ out) {
  int idx = blockIdx.x * blockDim.x + threadIdx.x;
  if (idx >= BS * D_MODEL) return;
  int d = idx & (D_MODEL - 1);
  int row = idx >> 8;
  float acc = bias[d];
  const float* r = dbl + (size_t)row * 48;
#pragma unroll
  for (int k = 0; k < DT_RANK; ++k) acc += r[k] * w[d * DT_RANK + k];
  out[idx] = softplus_f(acc);
}

// ------------------------------------------------- chunked selective scan, phase 1
__global__ void k_scan_partial(const float* __restrict__ dt0, const float* __restrict__ dt1,
                               const float* __restrict__ u0,  const float* __restrict__ u1,
                               const float* __restrict__ db0, const float* __restrict__ db1,
                               const float* __restrict__ Al0, const float* __restrict__ Al1,
                               float* __restrict__ hend, float* __restrict__ Pm) {
  int wid  = (blockIdx.x * blockDim.x + threadIdx.x) >> 5;
  int lane = threadIdx.x & 31;
  int c = wid & 63, dpair = (wid >> 6) & 127, b = (wid >> 13) & 1, dir = (wid >> 14) & 1;
  int d = dpair * 2 + (lane >> 4), n = lane & 15;
  const float* dt  = dir ? dt1 : dt0;
  const float* u   = dir ? u1  : u0;
  const float* dbl = dir ? db1 : db0;
  const float* Al  = dir ? Al1 : Al0;
  float A = -__expf(Al[d * D_STATE + n]);
  float h = 0.f, P = 1.f;
  int s0 = c * CHLEN;
  for (int sl = 0; sl < CHLEN; ++sl) {
    int s = s0 + sl;
    int t = dir ? (SEQ - 1 - s) : s;
    size_t row = (size_t)b * SEQ + t;
    float dtv = dt[row * D_MODEL + d];
    float uv  = u [row * D_MODEL + d];
    float Bv  = dbl[row * 48 + DT_RANK + n];
    float a = __expf(dtv * A);
    h = h * a + (dtv * uv) * Bv;
    P *= a;
  }
  size_t g = chunk_idx(dir, b, d, c, n);
  hend[g] = h;  Pm[g] = P;
}

// ------------------------------------------------- phase 2: serial carry over chunks
__global__ void k_scan_fix(const float* __restrict__ hend, const float* __restrict__ Pm,
                           float* __restrict__ Hst) {
  int wid  = (blockIdx.x * blockDim.x + threadIdx.x) >> 5;
  int lane = threadIdx.x & 31;
  int dpair = wid & 127, b = (wid >> 7) & 1, dir = (wid >> 8) & 1;
  int d = dpair * 2 + (lane >> 4), n = lane & 15;
  float H = 0.f;
  for (int c = 0; c < NCHUNK; ++c) {
    size_t g = chunk_idx(dir, b, d, c, n);
    Hst[g] = H;
    H = H * Pm[g] + hend[g];
  }
}

// ------------------------------------------------- phase 3: re-run chunks with carry-in, emit y
__global__ void k_scan_emit(const float* __restrict__ dt0, const float* __restrict__ dt1,
                            const float* __restrict__ u0,  const float* __restrict__ u1,
                            const float* __restrict__ db0, const float* __restrict__ db1,
                            const float* __restrict__ Al0, const float* __restrict__ Al1,
                            const float* __restrict__ Dp0, const float* __restrict__ Dp1,
                            const float* __restrict__ xz,  const float* __restrict__ Hst,
                            float* __restrict__ y0, float* __restrict__ y1) {
  int wid  = (blockIdx.x * blockDim.x + threadIdx.x) >> 5;
  int lane = threadIdx.x & 31;
  int c = wid & 63, dpair = (wid >> 6) & 127, b = (wid >> 13) & 1, dir = (wid >> 14) & 1;
  int d = dpair * 2 + (lane >> 4), n = lane & 15;
  const float* dt  = dir ? dt1 : dt0;
  const float* u   = dir ? u1  : u0;
  const float* dbl = dir ? db1 : db0;
  const float* Al  = dir ? Al1 : Al0;
  const float* Dpv = dir ? Dp1 : Dp0;
  float*       y   = dir ? y1  : y0;
  float A  = -__expf(Al[d * D_STATE + n]);
  float Dc = Dpv[d];
  float h  = Hst[chunk_idx(dir, b, d, c, n)];
  int s0 = c * CHLEN;
  for (int sl = 0; sl < CHLEN; ++sl) {
    int s = s0 + sl;
    int t = dir ? (SEQ - 1 - s) : s;
    size_t row = (size_t)b * SEQ + t;
    float dtv = dt[row * D_MODEL + d];
    float uv  = u [row * D_MODEL + d];
    float Bv  = dbl[row * 48 + DT_RANK + n];
    float Cv  = dbl[row * 48 + DT_RANK + D_STATE + n];
    float a = __expf(dtv * A);
    h = h * a + (dtv * uv) * Bv;
    float acc = half_sum16(h * Cv);
    if (n == 0) {
      float zv = xz[row * 512 + D_MODEL + d];
      y[row * D_MODEL + d] = (acc + uv * Dc) * silu_f(zv);
    }
  }
}

// ------------------------------------------------- y = 0.5*(yf+yb), LayerNorm -> f16
__global__ void k_combine_ln(const float* __restrict__ yf, const float* __restrict__ yb,
                             const float* __restrict__ g, const float* __restrict__ be,
                             _Float16* __restrict__ out, int rows) {
  int wid  = (blockIdx.x * blockDim.x + threadIdx.x) >> 5;
  int lane = threadIdx.x & 31;
  if (wid >= rows) return;
  size_t base = (size_t)wid * D_MODEL;
  float v[8]; float s = 0.f, s2 = 0.f;
#pragma unroll
  for (int i = 0; i < 8; ++i) {
    int c = lane * 8 + i;
    float t = 0.5f * (yf[base + c] + yb[base + c]);
    v[i] = t; s += t; s2 += t * t;
  }
  s = warp_sum32(s); s2 = warp_sum32(s2);
  float mean = s * (1.f / D_MODEL);
  float rstd = rsqrtf(s2 * (1.f / D_MODEL) - mean * mean + 1e-5f);
#pragma unroll
  for (int i = 0; i < 8; ++i) {
    int c = lane * 8 + i;
    out[base + c] = (_Float16)((v[i] - mean) * rstd * g[c] + be[c]);
  }
}

// ------------------------------------------------- post LN + residual -> d_out (f32)
__global__ void k_post_ln(const float* __restrict__ x, const float* __restrict__ g,
                          const float* __restrict__ be, const float* __restrict__ skip,
                          float* __restrict__ out, int rows) {
  int wid  = (blockIdx.x * blockDim.x + threadIdx.x) >> 5;
  int lane = threadIdx.x & 31;
  if (wid >= rows) return;
  size_t base = (size_t)wid * D_MODEL;
  float v[8]; float s = 0.f, s2 = 0.f;
#pragma unroll
  for (int i = 0; i < 8; ++i) { float t = x[base + lane * 8 + i]; v[i] = t; s += t; s2 += t * t; }
  s = warp_sum32(s); s2 = warp_sum32(s2);
  float mean = s * (1.f / D_MODEL);
  float rstd = rsqrtf(s2 * (1.f / D_MODEL) - mean * mean + 1e-5f);
#pragma unroll
  for (int i = 0; i < 8; ++i) {
    int c = lane * 8 + i;
    out[base + c] = (v[i] - mean) * rstd * g[c] + be[c] + skip[base + c];
  }
}

// ================================================================ host glue
extern "C" void kernel_launch(void* const* d_in, const int* in_sizes, int n_in,
                              void* d_out, int out_size, void* d_ws, size_t ws_size,
                              hipStream_t stream) {
  const float* input      = (const float*)d_in[0];
  const float* norm_g     = (const float*)d_in[1];
  const float* norm_b     = (const float*)d_in[2];
  const float* in_proj_w  = (const float*)d_in[3];
  const float* conv_w     = (const float*)d_in[4];
  const float* conv_b     = (const float*)d_in[5];
  const float* xproj_w    = (const float*)d_in[6];
  const float* dtproj_w   = (const float*)d_in[7];
  const float* dtproj_b   = (const float*)d_in[8];
  const float* A_log      = (const float*)d_in[9];
  const float* Dp         = (const float*)d_in[10];
  const float* conv_w_b   = (const float*)d_in[11];
  const float* conv_b_b   = (const float*)d_in[12];
  const float* xproj_w_b  = (const float*)d_in[13];
  const float* dtproj_w_b = (const float*)d_in[14];
  const float* dtproj_b_b = (const float*)d_in[15];
  const float* A_log_b    = (const float*)d_in[16];
  const float* Dp_b       = (const float*)d_in[17];
  const float* out_norm_g = (const float*)d_in[18];
  const float* out_norm_b = (const float*)d_in[19];
  const float* out_proj_w = (const float*)d_in[20];
  const float* post_norm_g= (const float*)d_in[21];
  const float* post_norm_b= (const float*)d_in[22];
  float* out = (float*)d_out;

  char* ws = (char*)d_ws;
  size_t off = 0;
  auto take = [&](size_t bytes) -> char* {
    char* p = ws + off;
    off = (off + bytes + 255) & ~(size_t)255;
    return p;
  };
  _Float16* xln_h  = (_Float16*)take((size_t)BS * 256 * 2);
  _Float16* win_h  = (_Float16*)take((size_t)512 * 256 * 2);
  _Float16* xpw0_h = (_Float16*)take((size_t)48 * 256 * 2);
  _Float16* xpw1_h = (_Float16*)take((size_t)48 * 256 * 2);
  _Float16* wout_h = (_Float16*)take((size_t)256 * 256 * 2);
  float*    xz     = (float*)   take((size_t)BS * 512 * 4);
  float*    u0     = (float*)   take((size_t)BS * 256 * 4);
  float*    u1     = (float*)   take((size_t)BS * 256 * 4);
  _Float16* u0h    = (_Float16*)take((size_t)BS * 256 * 2);
  _Float16* u1h    = (_Float16*)take((size_t)BS * 256 * 2);
  float*    dbl0   = (float*)   take((size_t)BS * 48 * 4);
  float*    dbl1   = (float*)   take((size_t)BS * 48 * 4);
  float*    dt0    = (float*)   take((size_t)BS * 256 * 4);
  float*    dt1    = (float*)   take((size_t)BS * 256 * 4);
  size_t nchunkstate = (size_t)2 * BATCH * D_MODEL * NCHUNK * 16;
  float*    hend   = (float*)   take(nchunkstate * 4);
  float*    Pm     = (float*)   take(nchunkstate * 4);
  float*    Hst    = (float*)   take(nchunkstate * 4);
  float*    y0     = (float*)   take((size_t)BS * 256 * 4);
  float*    y1     = (float*)   take((size_t)BS * 256 * 4);
  _Float16* ymid_h = (_Float16*)take((size_t)BS * 256 * 2);
  float*    otmp   = (float*)   take((size_t)BS * 256 * 4);

  auto blk = [](int threads) { return dim3((threads + TPB - 1) / TPB); };

  // 1) input LayerNorm -> f16
  k_ln_f16<<<blk(BS * 32), TPB, 0, stream>>>(input, norm_g, norm_b, xln_h, BS);

  // 2) weight conversions to f16
  k_cvt_f2h<<<blk(512 * 256), TPB, 0, stream>>>(in_proj_w, win_h, 512 * 256);
  k_cvt_f2h<<<blk(48 * 256),  TPB, 0, stream>>>(xproj_w,   xpw0_h, 48 * 256);
  k_cvt_f2h<<<blk(48 * 256),  TPB, 0, stream>>>(xproj_w_b, xpw1_h, 48 * 256);
  k_cvt_f2h<<<blk(256 * 256), TPB, 0, stream>>>(out_proj_w, wout_h, 256 * 256);

  // 3) in_proj: xz[8192,512] = xln @ Win^T  (TDM-staged WMMA, 128x64 macro-tiles)
  k_gemm_wmma_big<<<dim3((BS / 128) * (512 / 64)), TPB, 0, stream>>>(
      xln_h, win_h, xz, BS, 512);

  // 4) depthwise conv + SiLU, both directions
  k_conv_silu<<<blk(BS * 256), TPB, 0, stream>>>(
      xz, conv_w, conv_b, conv_w_b, conv_b_b, u0, u1, u0h, u1h);

  // 5) x_proj per direction: dbl[8192,48] = u @ Wx^T  (WMMA)
  k_gemm_wmma_small<<<blk((BS / 16) * (48 / 16) * 32), TPB, 0, stream>>>(
      u0h, xpw0_h, dbl0, BS, 48, 256);
  k_gemm_wmma_small<<<blk((BS / 16) * (48 / 16) * 32), TPB, 0, stream>>>(
      u1h, xpw1_h, dbl1, BS, 48, 256);

  // 6) dt projection + softplus per direction
  k_dtproj<<<blk(BS * 256), TPB, 0, stream>>>(dbl0, dtproj_w,   dtproj_b,   dt0);
  k_dtproj<<<blk(BS * 256), TPB, 0, stream>>>(dbl1, dtproj_w_b, dtproj_b_b, dt1);

  // 7) chunked selective scan: partial -> carry fix -> emit
  int scan_waves = 2 * BATCH * (D_MODEL / 2) * NCHUNK;       // 32768 waves
  k_scan_partial<<<blk(scan_waves * 32), TPB, 0, stream>>>(
      dt0, dt1, u0, u1, dbl0, dbl1, A_log, A_log_b, hend, Pm);
  int fix_waves = 2 * BATCH * (D_MODEL / 2);                 // 512 waves
  k_scan_fix<<<blk(fix_waves * 32), TPB, 0, stream>>>(hend, Pm, Hst);
  k_scan_emit<<<blk(scan_waves * 32), TPB, 0, stream>>>(
      dt0, dt1, u0, u1, dbl0, dbl1, A_log, A_log_b, Dp, Dp_b, xz, Hst, y0, y1);

  // 8) combine directions + out LayerNorm -> f16
  k_combine_ln<<<blk(BS * 32), TPB, 0, stream>>>(y0, y1, out_norm_g, out_norm_b, ymid_h, BS);

  // 9) out_proj: otmp[8192,256] = ymid @ Wout^T  (TDM-staged WMMA)
  k_gemm_wmma_big<<<dim3((BS / 128) * (256 / 64)), TPB, 0, stream>>>(
      ymid_h, wout_h, otmp, BS, 256);

  // 10) post LayerNorm + residual -> output (f32)
  k_post_ln<<<blk(BS * 32), TPB, 0, stream>>>(otmp, post_norm_g, post_norm_b, input, out, BS);
}
